// CausalSelfAttention_57973468561867
// MI455X (gfx1250) — compile-verified, matrix operand order checked by reading, not run on hardware
//
#include <hip/hip_runtime.h>
#include <hip/hip_bf16.h>

typedef __attribute__((ext_vector_type(16))) _Float16 v16h;
typedef __attribute__((ext_vector_type(8)))  _Float16 v8h;
typedef __attribute__((ext_vector_type(8)))  float    v8f;
typedef int v4i_vs __attribute__((vector_size(16)));

#define TSEQ   4096
#define NEMBD  1024
#define NHEAD  16
#define HDIM   64
#define BATCH  2
#define MROWS  (BATCH * TSEQ)   // 8192
#define NQKV   (3 * NEMBD)      // 3072

// ---- CDNA5 async global->LDS copy (ASYNCcnt-tracked), with fallback -------
#if defined(__has_builtin)
#  if __has_builtin(__builtin_amdgcn_global_load_async_to_lds_b128)
#    define ASYNC_LDS 1
#  endif
#endif
#ifndef ASYNC_LDS
#  define ASYNC_LDS 0
#endif

#if ASYNC_LDS
#warning CDNA5 probe: using __builtin_amdgcn_global_load_async_to_lds_b128
__device__ __forceinline__ void cp16_g2l(const _Float16* g, _Float16* l) {
  __builtin_amdgcn_global_load_async_to_lds_b128(
      (__attribute__((address_space(1))) v4i_vs*)(size_t)g,
      (__attribute__((address_space(3))) v4i_vs*)(unsigned)(size_t)l,
      0, 0);
}
#else
#warning CDNA5 probe: async-to-LDS builtin NOT available, using VGPR staging
#endif

// Load a 16x32 f16 WMMA fragment (A-pattern; B uses the same pattern on a
// transposed tile). Per CDNA5 ISA 16-bit A layout:
//   lanes 0-15 : row M=lane,    K = [0..7]  in v0..3, K=[16..23] in v4..7
//   lanes16-31 : row M=lane-16, K = [8..15] in v0..3, K=[24..31] in v4..7
__device__ __forceinline__ v16h frag_ld(const _Float16* base, int row, int rstride,
                                        int koff, int hf) {
  const _Float16* p = base + (size_t)row * rstride + koff + hf * 8;
  v8h lo = *(const v8h*)(p);
  v8h hi = *(const v8h*)(p + 16);
  return __builtin_shufflevector(lo, hi, 0,1,2,3,4,5,6,7,8,9,10,11,12,13,14,15);
}

__global__ void __launch_bounds__(256)
cvt_f32_f16(const float* __restrict__ s, _Float16* __restrict__ d, int n) {
  int i = blockIdx.x * 256 + threadIdx.x;
  if (i < n) d[i] = (_Float16)s[i];
}

// Transpose + convert weights once: W[K,N] f32 -> Wt[N,K] f16.
__global__ void __launch_bounds__(256)
cvt_transpose_f16(const float* __restrict__ W, _Float16* __restrict__ Wt,
                  int K, int N) {
  int idx = blockIdx.x * 256 + threadIdx.x;
  if (idx < K * N) {
    int k = idx / N, n = idx % N;
    Wt[(size_t)n * K + k] = (_Float16)W[idx];
  }
}

// ---------------------------------------------------------------------------
// Tiled f16 WMMA GEMM: C[M,N] = A[M,K] * Bt[N,K]^T + bias
// Double-buffered LDS tiles staged with global_load_async_to_lds_b128:
// tile i+1 streams in (ASYNCcnt) while WMMAs consume tile i; in-order async
// completion lets s_wait_asynccnt(8) drain exactly the previous tile.
// mode 0: write f32 to outd.
// mode 1: scatter QKV -> Q[b,h,t,d] f16, K[b,h,t,d] f16, V^T[b,h,d,t] f16.
// ---------------------------------------------------------------------------
__global__ void __launch_bounds__(256)
gemm_f16_wmma(const _Float16* __restrict__ A, const _Float16* __restrict__ Bt,
              const float* __restrict__ bias, int K, int N, int mode,
              _Float16* __restrict__ qd, _Float16* __restrict__ kd,
              _Float16* __restrict__ vtd, float* __restrict__ outd)
{
  __shared__ _Float16 As[2][128 * 64] __attribute__((aligned(16)));
  __shared__ _Float16 Bs[2][128 * 64] __attribute__((aligned(16)));  // [n][k]

  const int tid  = threadIdx.x;
  const int lane = tid & 31;
  const int hf   = lane >> 4;
  const int r    = lane & 15;
  const int wv   = tid >> 5;
  const int wm   = wv & 3;          // 4 wave-rows of 32
  const int wn   = wv >> 2;         // 2 wave-cols of 64
  const int m0   = blockIdx.y * 128;
  const int n0   = blockIdx.x * 128;

  // per-thread staging coordinates (8 chunks of 8 halves per row of 64)
  const int srow = tid >> 3;
  const int scol = (tid & 7) * 8;

#if ASYNC_LDS
  auto stage_async = [&](int it, int buf) {
    int kk = it * 64;
    #pragma unroll
    for (int c4 = 0; c4 < 4; ++c4) {
      int row = srow + c4 * 32;
      cp16_g2l(A  + (size_t)(m0 + row) * K + kk + scol, &As[buf][row * 64 + scol]);
      cp16_g2l(Bt + (size_t)(n0 + row) * K + kk + scol, &Bs[buf][row * 64 + scol]);
    }
  };
#else
  auto stage_sync = [&](int it, int buf) {
    int kk = it * 64;
    v8h areg[4], breg[4];
    #pragma unroll
    for (int c4 = 0; c4 < 4; ++c4) {
      int row = srow + c4 * 32;
      areg[c4] = *(const v8h*)(A  + (size_t)(m0 + row) * K + kk + scol);
      breg[c4] = *(const v8h*)(Bt + (size_t)(n0 + row) * K + kk + scol);
    }
    #pragma unroll
    for (int c4 = 0; c4 < 4; ++c4) {
      int row = srow + c4 * 32;
      *(v8h*)(&As[buf][row * 64 + scol]) = areg[c4];
      *(v8h*)(&Bs[buf][row * 64 + scol]) = breg[c4];
    }
  };
#endif

  v8f acc[2][4];
  #pragma unroll
  for (int i = 0; i < 2; ++i)
    #pragma unroll
    for (int j = 0; j < 4; ++j)
      #pragma unroll
      for (int e = 0; e < 8; ++e) acc[i][j][e] = 0.0f;

  const int NT = K / 64;
#if ASYNC_LDS
  stage_async(0, 0);                 // prologue: tile 0 -> buffer 0
#endif

  for (int it = 0; it < NT; ++it) {
    const int buf = it & 1;
#if ASYNC_LDS
    if (it + 1 < NT) {
      stage_async(it + 1, buf ^ 1);  // stream next tile into other buffer
      __builtin_amdgcn_s_wait_asynccnt(8);   // drain previous tile only
    } else {
      __builtin_amdgcn_s_wait_asynccnt(0);
    }
#else
    stage_sync(it, buf);
#endif
    __syncthreads();

    #pragma unroll
    for (int ks = 0; ks < 2; ++ks) {
      v16h af[2], bf[4];
      #pragma unroll
      for (int i = 0; i < 2; ++i)
        af[i] = frag_ld(As[buf], wm * 32 + i * 16 + r, 64, ks * 32, hf);
      #pragma unroll
      for (int j = 0; j < 4; ++j)
        bf[j] = frag_ld(Bs[buf], wn * 64 + j * 16 + r, 64, ks * 32, hf);
      #pragma unroll
      for (int i = 0; i < 2; ++i)
        #pragma unroll
        for (int j = 0; j < 4; ++j)
          acc[i][j] = __builtin_amdgcn_wmma_f32_16x16x32_f16(
              false, af[i], false, bf[j], (short)0, acc[i][j], false, false);
    }
    __syncthreads();
  }

  // Epilogue. C layout: VGPR rv, lanes 0-15 -> M=rv, lanes 16-31 -> M=rv+8; N=lane&15
  #pragma unroll
  for (int i = 0; i < 2; ++i)
    #pragma unroll
    for (int j = 0; j < 4; ++j)
      #pragma unroll
      for (int rv = 0; rv < 8; ++rv) {
        int grow = m0 + wm * 32 + i * 16 + rv + hf * 8;
        int gcol = n0 + wn * 64 + j * 16 + r;
        float v = acc[i][j][rv] + bias[gcol];
        if (mode == 0) {
          outd[(size_t)grow * N + gcol] = v;
        } else {
          int sel = gcol >> 10;       // 0:q 1:k 2:v
          int p   = gcol & 1023;
          int h   = p >> 6;
          int d   = p & 63;
          int b   = grow >> 12;
          int t   = grow & (TSEQ - 1);
          size_t hm = ((size_t)(b * NHEAD + h) * TSEQ + t) * HDIM + d;
          if (sel == 0)      qd[hm] = (_Float16)v;
          else if (sel == 1) kd[hm] = (_Float16)v;
          else vtd[((size_t)(b * NHEAD + h) * HDIM + d) * TSEQ + t] = (_Float16)v;
        }
      }
}

// ---------------------------------------------------------------------------
// In-place RoPE on Q and K (head-major f16). One thread per (b,h,t, pair j).
// ---------------------------------------------------------------------------
__global__ void __launch_bounds__(256)
rope_inplace(_Float16* __restrict__ Qd, _Float16* __restrict__ Kd)
{
  int idx = blockIdx.x * 256 + threadIdx.x;       // BATCH*NHEAD*TSEQ*32
  int j   = idx & 31;
  int t   = (idx >> 5) & (TSEQ - 1);
  int bh  = idx >> 17;
  size_t base = ((size_t)bh * TSEQ + t) * HDIM;
  float inv = __powf(10000.0f, -(float)(2 * j) / (float)HDIM);
  float ang = (float)t * inv;
  float c = __cosf(ang), s = __sinf(ang);
  float q1 = (float)Qd[base + j], q2 = (float)Qd[base + j + 32];
  Qd[base + j]      = (_Float16)(q1 * c - q2 * s);
  Qd[base + j + 32] = (_Float16)(q2 * c + q1 * s);
  float k1 = (float)Kd[base + j], k2 = (float)Kd[base + j + 32];
  Kd[base + j]      = (_Float16)(k1 * c - k2 * s);
  Kd[base + j + 32] = (_Float16)(k2 * c + k1 * s);
}

// ---------------------------------------------------------------------------
// Causal flash attention. Grid: BATCH*NHEAD*(TSEQ/64) blocks of 128 threads.
// The block cooperatively double-buffers the shared 64x64 K tile and 64x64
// V^T tile via async-to-LDS; each wave computes a 16-row Q tile with 16 WMMAs
// per 64-key online-softmax pass.
// ---------------------------------------------------------------------------
__global__ void __launch_bounds__(128)
flash_attn(const _Float16* __restrict__ Qd, const _Float16* __restrict__ Kd,
           const _Float16* __restrict__ Vtd, _Float16* __restrict__ Od)
{
  __shared__ _Float16 Ks[2][64 * 64] __attribute__((aligned(16)));  // [key][d]
  __shared__ _Float16 Vs[2][64 * 64] __attribute__((aligned(16)));  // [d][key]
  // 16 rows x (64 keys + 8 pad) halves per wave; row stride 144B (16B-aligned)
  __shared__ _Float16 pbuf[4][16 * 72] __attribute__((aligned(16)));

  const int tid  = threadIdx.x;
  const int lane = tid & 31;
  const int wave = tid >> 5;
  const int hf   = lane >> 4;
  const int r    = lane & 15;

  const int qb = blockIdx.x & 63;      // TSEQ/64 q-blocks
  const int bh = blockIdx.x >> 6;      // 0..BATCH*NHEAD-1
  const int b  = bh >> 4;
  const int h  = bh & 15;
  const int q0 = qb * 64 + wave * 16;

  const _Float16* Qh  = Qd  + (size_t)bh * TSEQ * HDIM;
  const _Float16* Kh  = Kd  + (size_t)bh * TSEQ * HDIM;
  const _Float16* Vth = Vtd + (size_t)bh * HDIM * TSEQ;

  // staging coordinates: 64x64 tile = 512 16B chunks, 4 per thread
  const int crow = tid >> 3;           // 0..15
  const int ccol = (tid & 7) * 8;      // 0..56

#if ASYNC_LDS
  auto stage_kv = [&](int kb_, int buf) {
    int kB_ = kb_ * 64;
    #pragma unroll
    for (int i = 0; i < 4; ++i) {
      int row = crow + i * 16;
      cp16_g2l(Kh  + (size_t)(kB_ + row) * HDIM + ccol, &Ks[buf][row * 64 + ccol]);
      cp16_g2l(Vth + (size_t)row * TSEQ + kB_ + ccol,   &Vs[buf][row * 64 + ccol]);
    }
  };
#else
  auto stage_kv_sync = [&](int kb_, int buf) {
    int kB_ = kb_ * 64;
    v8h kreg[4], vreg[4];
    #pragma unroll
    for (int i = 0; i < 4; ++i) {
      int row = crow + i * 16;
      kreg[i] = *(const v8h*)(Kh  + (size_t)(kB_ + row) * HDIM + ccol);
      vreg[i] = *(const v8h*)(Vth + (size_t)row * TSEQ + kB_ + ccol);
    }
    #pragma unroll
    for (int i = 0; i < 4; ++i) {
      int row = crow + i * 16;
      *(v8h*)(&Ks[buf][row * 64 + ccol]) = kreg[i];
      *(v8h*)(&Vs[buf][row * 64 + ccol]) = vreg[i];
    }
  };
#endif

  v16h aq[2];
  #pragma unroll
  for (int ks = 0; ks < 2; ++ks)
    aq[ks] = frag_ld(Qh, q0 + r, HDIM, ks * 32, hf);

  float mrow[8], lrow[8];
  v8f o[4];
  #pragma unroll
  for (int rv = 0; rv < 8; ++rv) { mrow[rv] = -1e30f; lrow[rv] = 0.0f; }
  #pragma unroll
  for (int tn = 0; tn < 4; ++tn)
    #pragma unroll
    for (int e = 0; e < 8; ++e) o[tn][e] = 0.0f;

  const float scale = 0.125f;              // 1/sqrt(64)
  const int nkb = qb + 1;                  // 64-key blocks; uniform over waves

#if ASYNC_LDS
  stage_kv(0, 0);                          // prologue
#endif

  for (int kb = 0; kb < nkb; ++kb) {
    const int kB  = kb * 64;
    const int buf = kb & 1;
#if ASYNC_LDS
    if (kb + 1 < nkb) {
      stage_kv(kb + 1, buf ^ 1);
      __builtin_amdgcn_s_wait_asynccnt(8);
    } else {
      __builtin_amdgcn_s_wait_asynccnt(0);
    }
#else
    stage_kv_sync(kb, buf);
#endif
    __syncthreads();

    // ---- S = Q @ K^T for 64 keys (8 WMMAs) ----
    v8f S[4];
    #pragma unroll
    for (int tn = 0; tn < 4; ++tn)
      #pragma unroll
      for (int e = 0; e < 8; ++e) S[tn][e] = 0.0f;

    #pragma unroll
    for (int tn = 0; tn < 4; ++tn)
      #pragma unroll
      for (int ks = 0; ks < 2; ++ks) {
        v16h bk = frag_ld(Ks[buf], tn * 16 + r, 64, ks * 32, hf);
        S[tn] = __builtin_amdgcn_wmma_f32_16x16x32_f16(
            false, aq[ks], false, bk, (short)0, S[tn], false, false);
      }

    // ---- scale + causal mask + online softmax (row = rv + hf*8, col = r) ----
    #pragma unroll
    for (int rv = 0; rv < 8; ++rv) {
      int qrow = q0 + rv + hf * 8;
      float s[4];
      #pragma unroll
      for (int tn = 0; tn < 4; ++tn) {
        s[tn] = S[tn][rv] * scale;
        if (kB + tn * 16 + r > qrow) s[tn] = -1e30f;
      }
      float rm = fmaxf(fmaxf(s[0], s[1]), fmaxf(s[2], s[3]));
      #pragma unroll
      for (int off = 1; off < 16; off <<= 1)
        rm = fmaxf(rm, __shfl_xor(rm, off, 16));
      float mnew = fmaxf(mrow[rv], rm);
      float p[4];
      #pragma unroll
      for (int tn = 0; tn < 4; ++tn) p[tn] = __expf(s[tn] - mnew);
      float rs = (p[0] + p[1]) + (p[2] + p[3]);
      #pragma unroll
      for (int off = 1; off < 16; off <<= 1)
        rs += __shfl_xor(rs, off, 16);
      float alpha = __expf(mrow[rv] - mnew);
      lrow[rv] = lrow[rv] * alpha + rs;
      mrow[rv] = mnew;
      #pragma unroll
      for (int tn = 0; tn < 4; ++tn) o[tn][rv] *= alpha;
      int Mr = rv + hf * 8;
      #pragma unroll
      for (int tn = 0; tn < 4; ++tn)
        pbuf[wave][Mr * 72 + tn * 16 + r] = (_Float16)p[tn];
    }
    __syncthreads();

    // ---- O += P @ V for 64 keys (8 WMMAs) ----
    v16h pa[2];
    #pragma unroll
    for (int ks = 0; ks < 2; ++ks)
      pa[ks] = frag_ld(pbuf[wave], r, 72, ks * 32, hf);
    #pragma unroll
    for (int tn = 0; tn < 4; ++tn)
      #pragma unroll
      for (int ks = 0; ks < 2; ++ks) {
        v16h bv = frag_ld(Vs[buf], tn * 16 + r, 64, ks * 32, hf);
        o[tn] = __builtin_amdgcn_wmma_f32_16x16x32_f16(
            false, pa[ks], false, bv, (short)0, o[tn], false, false);
      }
    __syncthreads();
  }

  #pragma unroll
  for (int tn = 0; tn < 4; ++tn)
    #pragma unroll
    for (int rv = 0; rv < 8; ++rv) {
      int t = q0 + rv + hf * 8;
      float v = o[tn][rv] / lrow[rv];
      Od[((size_t)(b * TSEQ + t)) * NEMBD + h * HDIM + tn * 16 + r] = (_Float16)v;
    }
}

// ---------------------------------------------------------------------------
extern "C" void kernel_launch(void* const* d_in, const int* in_sizes, int n_in,
                              void* d_out, int out_size, void* d_ws, size_t ws_size,
                              hipStream_t stream)
{
  const float* x      = (const float*)d_in[0];
  const float* W_attn = (const float*)d_in[1];
  const float* b_attn = (const float*)d_in[2];
  const float* W_proj = (const float*)d_in[3];
  const float* b_proj = (const float*)d_in[4];
  float* out = (float*)d_out;

  char* ws = (char*)d_ws;
  size_t off = 0;
  auto alloc = [&](size_t bytes) {
    char* p = ws + off;
    off += (bytes + 255) & ~(size_t)255;
    return p;
  };
  _Float16* xh     = (_Float16*)alloc((size_t)MROWS * NEMBD * 2);          // 16 MB
  _Float16* wqkvt  = (_Float16*)alloc((size_t)NQKV * NEMBD * 2);           //  6 MB [N,K]
  _Float16* wprojt = (_Float16*)alloc((size_t)NEMBD * NEMBD * 2);          //  2 MB [N,K]
  _Float16* Qb     = (_Float16*)alloc((size_t)BATCH * NHEAD * TSEQ * HDIM * 2); // 16 MB
  _Float16* Kb     = (_Float16*)alloc((size_t)BATCH * NHEAD * TSEQ * HDIM * 2); // 16 MB
  _Float16* Vtb    = (_Float16*)alloc((size_t)BATCH * NHEAD * HDIM * TSEQ * 2); // 16 MB
  _Float16* Ob     = (_Float16*)alloc((size_t)MROWS * NEMBD * 2);          // 16 MB

  // 1. f32 -> f16 conversions; weights transposed to [N,K] once
  cvt_f32_f16<<<(MROWS * NEMBD) / 256, 256, 0, stream>>>(x, xh, MROWS * NEMBD);
  cvt_transpose_f16<<<(NEMBD * NQKV) / 256, 256, 0, stream>>>(W_attn, wqkvt, NEMBD, NQKV);
  cvt_transpose_f16<<<(NEMBD * NEMBD) / 256, 256, 0, stream>>>(W_proj, wprojt, NEMBD, NEMBD);

  // 2. QKV GEMM + bias, scatter to head-major Q, K, V^T
  gemm_f16_wmma<<<dim3(NQKV / 128, MROWS / 128), 256, 0, stream>>>(
      xh, wqkvt, b_attn, NEMBD, NQKV, 1, Qb, Kb, Vtb, nullptr);

  // 3. RoPE in place on Q, K
  rope_inplace<<<(BATCH * NHEAD * TSEQ * 32) / 256, 256, 0, stream>>>(Qb, Kb);

  // 4. Causal flash attention -> O (f16, [B,T,C])
  flash_attn<<<BATCH * NHEAD * (TSEQ / 64), 128, 0, stream>>>(Qb, Kb, Vtb, Ob);

  // 5. Output projection + bias -> f32 out
  gemm_f16_wmma<<<dim3(NEMBD / 128, MROWS / 128), 256, 0, stream>>>(
      Ob, wprojt, b_proj, NEMBD, NEMBD, 0, nullptr, nullptr, nullptr, out);
}
